// RoutingFreeMaskedMoE_72438918414735
// MI455X (gfx1250) — matrix operand order repaired and checked by reading, not compiled, not run on hardware
//
#include <hip/hip_runtime.h>
#include <hip/hip_bf16.h>

// ---------------- problem constants (from reference) ----------------
#define NTOK 8192   // B*T = 4*2048
#define DD   1024   // hidden dim
#define EE   8      // experts
#define RR   8      // gate rank
#define FF   1024   // ffn dim
#define FNT  128    // N-tile (f / d columns per workgroup)

typedef __attribute__((ext_vector_type(16))) __bf16 v16bf;
typedef __attribute__((ext_vector_type(8)))  float  v8f;

union ABFrag { uint4 q[2]; v16bf v; };   // 32 bytes = 8 VGPRs per lane
union Pack4  { __bf16 h[4]; uint2 u; };  // 4 bf16 -> 8B store

// ---------------- CDNA5 async global->LDS staging ----------------
#if defined(__has_builtin)
#if __has_builtin(__builtin_amdgcn_global_load_async_to_lds_b128)
#define HAVE_ASYNC_LDS 1
#endif
#if __has_builtin(__builtin_amdgcn_s_wait_asynccnt)
#define HAVE_WAIT_ASYNC 1
#endif
#endif
#ifndef HAVE_ASYNC_LDS
#define HAVE_ASYNC_LDS 0
#endif
#ifndef HAVE_WAIT_ASYNC
#define HAVE_WAIT_ASYNC 0
#endif

// pointer types matching the builtin's signature:
//   (addrspace(1) v4i* src, addrspace(3) v4i* dst, imm offset, imm cpol)
typedef int v4i_gcc __attribute__((vector_size(16)));
typedef __attribute__((address_space(1))) v4i_gcc as1_v4i;
typedef __attribute__((address_space(3))) v4i_gcc as3_v4i;

// copy 16B global -> LDS (async on CDNA5: global_load_async_to_lds_b128)
__device__ inline void stage16(__bf16* dst_lds, const __bf16* src) {
#if HAVE_ASYNC_LDS
  __builtin_amdgcn_global_load_async_to_lds_b128(
      (as1_v4i*)src, (as3_v4i*)dst_lds, 0, 0);
#else
  *reinterpret_cast<uint4*>(dst_lds) = *reinterpret_cast<const uint4*>(src);
#endif
}

__device__ inline void wait_stage() {
#if HAVE_ASYNC_LDS
#if HAVE_WAIT_ASYNC
  __builtin_amdgcn_s_wait_asynccnt(0);
#else
  asm volatile("s_wait_asynccnt 0x0" ::: "memory");
#endif
#endif
}

// ---------------- workspace layout (bytes) ----------------
#define WS_XB  ((size_t)0)                         // x bf16:  N*D*2  = 16 MB
#define WS_W   (WS_XB + (size_t)NTOK*DD*2)         // weights: N*E*4  = 256 KB
#define WS_WG  (WS_W  + (size_t)NTOK*EE*4)         // Wg bf16: F*D*2  = 2 MB
#define WS_WU  (WS_WG + (size_t)FF*DD*2)           // Wu bf16: 2 MB
#define WS_WD  (WS_WU + (size_t)FF*DD*2)           // Wd bf16: 2 MB
#define WS_H   (WS_WD + (size_t)DD*FF*2)           // H bf16:  N*F*2  = 16 MB

// ==================================================================
// Gate: per token n, h[e,r] = x . W_A[e,r,:]; score = sqrt(mean_r h^2 + eps)
// mask = score >= 1; also emits x converted to bf16 for the WMMA GEMMs.
// block = 64 threads (2 waves), one token per block; thread t = e*8+r.
// ==================================================================
__global__ __launch_bounds__(64) void gate_kernel(
    const float* __restrict__ x, const float* __restrict__ WA,
    __bf16* __restrict__ xb, float* __restrict__ wsw,
    float* __restrict__ scores)
{
  __shared__ __align__(16) float xrow[DD];
  __shared__ float acc2[64];
  const int n = blockIdx.x;
  const int t = threadIdx.x;
  const float* xp = x + (size_t)n * DD;

  const int base = t * 16;
  #pragma unroll
  for (int i = 0; i < 16; i += 4) {
    float4 v = *reinterpret_cast<const float4*>(&xp[base + i]);
    xrow[base + i + 0] = v.x; xrow[base + i + 1] = v.y;
    xrow[base + i + 2] = v.z; xrow[base + i + 3] = v.w;
    Pack4 p;
    p.h[0] = (__bf16)v.x; p.h[1] = (__bf16)v.y;
    p.h[2] = (__bf16)v.z; p.h[3] = (__bf16)v.w;
    *reinterpret_cast<uint2*>(&xb[(size_t)n * DD + base + i]) = p.u;
  }
  __syncthreads();

  const float* wa = WA + (size_t)t * DD;   // thread t = gate row (e,r)
  float acc = 0.0f;
  for (int d = 0; d < DD; d += 4) {
    float4 xv = *reinterpret_cast<const float4*>(&xrow[d]);
    float4 wv = *reinterpret_cast<const float4*>(&wa[d]);
    acc = fmaf(xv.x, wv.x, acc);
    acc = fmaf(xv.y, wv.y, acc);
    acc = fmaf(xv.z, wv.z, acc);
    acc = fmaf(xv.w, wv.w, acc);
  }
  acc2[t] = acc * acc;
  __syncthreads();

  if (t < EE) {
    float s = 0.0f;
    #pragma unroll
    for (int r = 0; r < RR; ++r) s += acc2[t * RR + r];
    // temperature=1, scale=1, bias=0, threshold=1
    float score = sqrtf(s * (1.0f / RR) + 1e-6f);
    bool m = score >= 1.0f;
    wsw[(size_t)n * EE + t]    = m ? score : 0.0f;
    scores[(size_t)n * EE + t] = m ? score : __uint_as_float(0xff800000u);
  }
}

// ==================================================================
// Convert one expert's Wg/Wu ([F,D]) and Wd ([D,F]) f32 -> bf16 in ws.
// ==================================================================
__global__ __launch_bounds__(256) void cvt_kernel(
    const float* __restrict__ wg, const float* __restrict__ wu,
    const float* __restrict__ wd,
    __bf16* __restrict__ wgb, __bf16* __restrict__ wub,
    __bf16* __restrict__ wdb)
{
  const size_t i = ((size_t)blockIdx.x * 256 + threadIdx.x) * 4;
  float4 g = *reinterpret_cast<const float4*>(&wg[i]);
  float4 u = *reinterpret_cast<const float4*>(&wu[i]);
  float4 d = *reinterpret_cast<const float4*>(&wd[i]);
  Pack4 pg, pu, pd;
  pg.h[0]=(__bf16)g.x; pg.h[1]=(__bf16)g.y; pg.h[2]=(__bf16)g.z; pg.h[3]=(__bf16)g.w;
  pu.h[0]=(__bf16)u.x; pu.h[1]=(__bf16)u.y; pu.h[2]=(__bf16)u.z; pu.h[3]=(__bf16)u.w;
  pd.h[0]=(__bf16)d.x; pd.h[1]=(__bf16)d.y; pd.h[2]=(__bf16)d.z; pd.h[3]=(__bf16)d.w;
  *reinterpret_cast<uint2*>(&wgb[i]) = pg.u;
  *reinterpret_cast<uint2*>(&wub[i]) = pu.u;
  *reinterpret_cast<uint2*>(&wdb[i]) = pd.u;
}

// ------------------------------------------------------------------
// WMMA fragment helpers (A/B in this file's contiguous k order, not the ISA table order; wave32)
// A (16x32 bf16, M=lane&15):
//   lanes 0-15 : elems 0..15 = K 0..15
//   lanes 16-31: elems 0..15 = K 16..31
// B (32x16 bf16, N=lane&15): elems 0..15 = K koff..koff+15, koff=(lane>>4)*16
// C/D (16x16 f32): elem i -> row M = i + 8*(lane>>4), col N = lane&15
// ------------------------------------------------------------------
__device__ inline v16bf load_a_frag(const __bf16* __restrict__ row_base,
                                    int khalf)
{
  ABFrag a;
  const uint4* ap = reinterpret_cast<const uint4*>(row_base + 2 * khalf);
  a.q[0] = ap[0];   // K 2*khalf .. 2*khalf+7
  a.q[1] = ap[1];   // K 2*khalf+8 .. 2*khalf+15
  return a.v;
}

__device__ inline v16bf load_b_frag(const __bf16* lds_row, int koff)
{
  ABFrag b;
  const uint4* bp = reinterpret_cast<const uint4*>(lds_row + koff);
  b.q[0] = bp[0];
  b.q[1] = bp[1];
  return b.v;
}

// ==================================================================
// FFN stage 1 (per expert): H[n,f] = silu(x.Wg^T) * (x.Wu^T), bf16 out.
// WG = 8 waves: 128 rows x 128 cols; wave w -> rows [m0+16w, +16).
// Weight k-chunks ([128 rows][32 k] bf16, 8KB each) double-buffered in LDS,
// staged with async global->LDS copies.
// ==================================================================
__global__ __launch_bounds__(256) void ffn1_kernel(
    const __bf16* __restrict__ xb, const __bf16* __restrict__ wgb,
    const __bf16* __restrict__ wub, __bf16* __restrict__ hb)
{
  __shared__ __align__(16) __bf16 lds_wg[2][FNT * 32];
  __shared__ __align__(16) __bf16 lds_wu[2][FNT * 32];

  const int m0 = blockIdx.x * 128;
  const int f0 = blockIdx.y * FNT;
  const int t = threadIdx.x;
  const int lane = t & 31;
  const int wv = t >> 5;
  const int mw = m0 + wv * 16;

  v8f Gacc[8], Uacc[8];
  #pragma unroll
  for (int j = 0; j < 8; ++j) {
    Gacc[j] = (v8f){0.f,0.f,0.f,0.f,0.f,0.f,0.f,0.f};
    Uacc[j] = (v8f){0.f,0.f,0.f,0.f,0.f,0.f,0.f,0.f};
  }

  const int srow = t >> 1;          // 0..127 : staged weight row
  const int sq   = (t & 1) * 16;    // 0 / 16 : k sub-offset (elems)
  const int nc   = lane & 15;
  const int khalf = (lane >> 4) * 8;
  const int koffB = (lane >> 4) * 16;
  const __bf16* a_row = xb + (size_t)(mw + nc) * DD;

  const size_t grow = (size_t)(f0 + srow) * DD + sq;
  const int sl = srow * 32 + sq;

  // prologue: stage chunk 0 into buffer 0
  stage16(&lds_wg[0][sl],     &wgb[grow]);
  stage16(&lds_wg[0][sl + 8], &wgb[grow + 8]);
  stage16(&lds_wu[0][sl],     &wub[grow]);
  stage16(&lds_wu[0][sl + 8], &wub[grow + 8]);
  wait_stage();
  __syncthreads();

  int buf = 0;
  for (int kc = 0; kc < DD; kc += 32) {
    if (kc + 32 < DD) {           // stage next chunk into the other buffer
      const size_t g = grow + kc + 32;
      stage16(&lds_wg[buf ^ 1][sl],     &wgb[g]);
      stage16(&lds_wg[buf ^ 1][sl + 8], &wgb[g + 8]);
      stage16(&lds_wu[buf ^ 1][sl],     &wub[g]);
      stage16(&lds_wu[buf ^ 1][sl + 8], &wub[g + 8]);
      if (kc + 64 < DD) {         // pull chunk k+2 toward L2 (global_prefetch_b8)
        __builtin_prefetch(&wgb[g + 32], 0, 1);
        __builtin_prefetch(&wub[g + 32], 0, 1);
      }
    }

    v16bf a = load_a_frag(a_row + kc, khalf);
    #pragma unroll
    for (int j = 0; j < 8; ++j) {
      v16bf bg = load_b_frag(&lds_wg[buf][(j * 16 + nc) * 32], koffB);
      Gacc[j] = __builtin_amdgcn_wmma_f32_16x16x32_bf16(
          false, a, false, bg, (short)0, Gacc[j], false, false);
      v16bf bu = load_b_frag(&lds_wu[buf][(j * 16 + nc) * 32], koffB);
      Uacc[j] = __builtin_amdgcn_wmma_f32_16x16x32_bf16(
          false, a, false, bu, (short)0, Uacc[j], false, false);
    }

    wait_stage();
    __syncthreads();
    buf ^= 1;
  }

  // epilogue: H = silu(G) * U -> bf16
  const int mh = (lane >> 4) * 8;
  #pragma unroll
  for (int j = 0; j < 8; ++j) {
    #pragma unroll
    for (int i = 0; i < 8; ++i) {
      float g = Gacc[j][i];
      float u = Uacc[j][i];
      float h = (g / (1.0f + __expf(-g))) * u;
      hb[(size_t)(mw + mh + i) * FF + f0 + j * 16 + nc] = (__bf16)h;
    }
  }
}

// ==================================================================
// FFN stage 2 (per expert): out[n,d] (+)= w[n,e] * sum_f H[n,f]*Wd[d,f]
// Same 128x128 tiling; K = F. e==0 overwrites out (no zero-init pass).
// ==================================================================
__global__ __launch_bounds__(256) void ffn2_kernel(
    const __bf16* __restrict__ hb, const __bf16* __restrict__ wdb,
    const float* __restrict__ wsw, float* __restrict__ out,
    int e, int first)
{
  __shared__ __align__(16) __bf16 lds_wd[2][FNT * 32];

  const int m0 = blockIdx.x * 128;
  const int d0 = blockIdx.y * FNT;
  const int t = threadIdx.x;
  const int lane = t & 31;
  const int wv = t >> 5;
  const int mw = m0 + wv * 16;

  v8f acc[8];
  #pragma unroll
  for (int j = 0; j < 8; ++j)
    acc[j] = (v8f){0.f,0.f,0.f,0.f,0.f,0.f,0.f,0.f};

  const int srow = t >> 1;
  const int sq   = (t & 1) * 16;
  const int nc   = lane & 15;
  const int khalf = (lane >> 4) * 8;
  const int koffB = (lane >> 4) * 16;
  const __bf16* a_row = hb + (size_t)(mw + nc) * FF;

  const size_t grow = (size_t)(d0 + srow) * FF + sq;
  const int sl = srow * 32 + sq;

  stage16(&lds_wd[0][sl],     &wdb[grow]);
  stage16(&lds_wd[0][sl + 8], &wdb[grow + 8]);
  wait_stage();
  __syncthreads();

  int buf = 0;
  for (int kc = 0; kc < FF; kc += 32) {
    if (kc + 32 < FF) {
      const size_t g = grow + kc + 32;
      stage16(&lds_wd[buf ^ 1][sl],     &wdb[g]);
      stage16(&lds_wd[buf ^ 1][sl + 8], &wdb[g + 8]);
      if (kc + 64 < FF)
        __builtin_prefetch(&wdb[g + 32], 0, 1);
    }

    v16bf a = load_a_frag(a_row + kc, khalf);
    #pragma unroll
    for (int j = 0; j < 8; ++j) {
      v16bf b = load_b_frag(&lds_wd[buf][(j * 16 + nc) * 32], koffB);
      acc[j] = __builtin_amdgcn_wmma_f32_16x16x32_bf16(
          false, a, false, b, (short)0, acc[j], false, false);
    }

    wait_stage();
    __syncthreads();
    buf ^= 1;
  }

  const int mh = (lane >> 4) * 8;
  float wgt[8];
  #pragma unroll
  for (int i = 0; i < 8; ++i)
    wgt[i] = wsw[(size_t)(mw + mh + i) * EE + e];

  #pragma unroll
  for (int j = 0; j < 8; ++j) {
    #pragma unroll
    for (int i = 0; i < 8; ++i) {
      const size_t idx = (size_t)(mw + mh + i) * DD + d0 + j * 16 + nc;
      float v = acc[j][i] * wgt[i];
      if (first) out[idx] = v;
      else       out[idx] += v;
    }
  }
}

// ==================================================================
extern "C" void kernel_launch(void* const* d_in, const int* in_sizes, int n_in,
                              void* d_out, int out_size, void* d_ws, size_t ws_size,
                              hipStream_t stream) {
  const float* x      = (const float*)d_in[0];   // [N, D]
  const float* W_A    = (const float*)d_in[1];   // [E, R, D]
  const float* W_gate = (const float*)d_in[2];   // [E, F, D]
  const float* W_up   = (const float*)d_in[3];   // [E, F, D]
  const float* W_down = (const float*)d_in[4];   // [E, D, F]

  float* out    = (float*)d_out;                 // [N, D]
  float* scores = out + (size_t)NTOK * DD;       // [N, E]

  char* ws = (char*)d_ws;
  __bf16* xb  = (__bf16*)(ws + WS_XB);
  float*  wsw = (float*) (ws + WS_W);
  __bf16* wgb = (__bf16*)(ws + WS_WG);
  __bf16* wub = (__bf16*)(ws + WS_WU);
  __bf16* wdb = (__bf16*)(ws + WS_WD);
  __bf16* hb  = (__bf16*)(ws + WS_H);

  // 1) gate scores/weights + x -> bf16
  gate_kernel<<<NTOK, 64, 0, stream>>>(x, W_A, xb, wsw, scores);

  // 2) per expert: convert weights, FFN1 (H), FFN2 (accumulate out)
  const dim3 gemm_grid(NTOK / 128, FF / FNT);
  for (int e = 0; e < EE; ++e) {
    cvt_kernel<<<(FF * DD) / (256 * 4), 256, 0, stream>>>(
        W_gate + (size_t)e * FF * DD, W_up + (size_t)e * FF * DD,
        W_down + (size_t)e * DD * FF, wgb, wub, wdb);
    ffn1_kernel<<<gemm_grid, 256, 0, stream>>>(xb, wgb, wub, hb);
    ffn2_kernel<<<gemm_grid, 256, 0, stream>>>(hb, wdb, wsw, out, e, e == 0);
  }
}
